// LinearGOATLayer_74156905333519
// MI455X (gfx1250) — matrix operand order repaired and checked by reading, not compiled, not run on hardware
//
#include <hip/hip_runtime.h>
#include <math.h>
#include <stdint.h>

#define DIN   4096
#define DOUT  4096
#define LDSTR 40                  // ushorts per 32-K row (+8 pad) -> 80B stride
#define PLANE_USH (128 * LDSTR)   // 5120 ushorts = 10240 B per LDS plane
#define TILE_B (4 * PLANE_USH * 2)  // 40960 B per pipeline stage (4 planes)

typedef __attribute__((ext_vector_type(16))) __bf16         v16bf;
typedef __attribute__((ext_vector_type(8)))  float          v8f;
typedef __attribute__((ext_vector_type(2)))  float          v2f;
typedef __attribute__((ext_vector_type(2)))  unsigned short us2;
typedef __attribute__((ext_vector_type(4)))  unsigned short us4;
typedef __attribute__((ext_vector_type(8)))  unsigned short us8;
typedef __attribute__((ext_vector_type(16))) unsigned short us16;

// ---- fp32 -> (bf16 hi, bf16 lo) round-to-nearest-even split ----------------
__device__ __forceinline__ unsigned short bf16_rne(float x) {
  unsigned int u = __builtin_bit_cast(unsigned int, x);
  u += 0x7FFFu + ((u >> 16) & 1u);
  return (unsigned short)(u >> 16);
}
__device__ __forceinline__ void split2(float x, unsigned short& h, unsigned short& l) {
  unsigned short hs = bf16_rne(x);
  float hf = __builtin_bit_cast(float, (unsigned int)hs << 16);
  h = hs;
  l = bf16_rne(x - hf);
}
__device__ __forceinline__ void split4(float4 v, us4& h, us4& l) {
  unsigned short h0, l0, h1, l1, h2, l2, h3, l3;
  split2(v.x, h0, l0); split2(v.y, h1, l1);
  split2(v.z, h2, l2); split2(v.w, h3, l3);
  h[0] = h0; h[1] = h1; h[2] = h2; h[3] = h3;
  l[0] = l0; l[1] = l1; l[2] = l2; l[3] = l3;
}

// ---- WMMA fragment loaders from LDS (bf16, 16x32 A / 32x16 B) --------------
__device__ __forceinline__ v16bf fragA(const unsigned short* plane, int row0, int lane) {
  int l2 = lane & 15, s = (lane >> 4) & 1;
  const unsigned short* p = plane + (row0 + l2) * LDSTR + s * 8;
  us8 a = *(const us8*)p;          // K s*8 .. s*8+7
  us8 b = *(const us8*)(p + 16);   // K s*8+16 .. s*8+23
  us16 c = __builtin_shufflevector(a, b, 0,1,2,3,4,5,6,7,8,9,10,11,12,13,14,15);
  return __builtin_bit_cast(v16bf, c);
}
__device__ __forceinline__ v16bf fragB(const unsigned short* plane, int row0, int lane) {
  int l2 = lane & 15, s = (lane >> 4) & 1;
  const unsigned short* p = plane + (row0 + l2) * LDSTR + s * 8;
  us8 a = *(const us8*)p;
  us8 b = *(const us8*)(p + 16);
  us16 c = __builtin_shufflevector(a, b, 0,1,2,3,4,5,6,7,8,9,10,11,12,13,14,15);
  return __builtin_bit_cast(v16bf, c);
}
__device__ __forceinline__ v8f vzero8() {
  v8f z = {0.f,0.f,0.f,0.f,0.f,0.f,0.f,0.f};
  return z;
}

// ---- CDNA5 async global->LDS copy (ASYNCcnt), GVS mode ---------------------
__device__ __forceinline__ void async_ld_b128(unsigned lds_off, unsigned voff,
                                              const void* sbase) {
  asm volatile("global_load_async_to_lds_b128 %0, %1, %2"
               :: "v"(lds_off), "v"(voff), "s"(sbase) : "memory");
}
__device__ __forceinline__ void wait_async0()  { asm volatile("s_wait_asynccnt 0x0" ::: "memory"); }
__device__ __forceinline__ void wait_async8()  { asm volatile("s_wait_asynccnt 0x8" ::: "memory"); }
__device__ __forceinline__ void wait_async16() { asm volatile("s_wait_asynccnt 0x10" ::: "memory"); }

// ============================================================================
// Prep: elementwise fp32 -> bf16 hi/lo planes (one conversion per element).
// ============================================================================
__global__ __launch_bounds__(256) void split_planes_kernel(
    const float* __restrict__ src, unsigned short* __restrict__ hi,
    unsigned short* __restrict__ lo, int n4)
{
  int i = blockIdx.x * 256 + threadIdx.x;
  if (i < n4) {
    float4 v = ((const float4*)src)[i];
    us4 h, l;
    split4(v, h, l);
    ((us4*)hi)[i] = h;
    ((us4*)lo)[i] = l;
  }
}

// ============================================================================
// Kernel 1: gate logits + LoRA-A projection as one bf16x3 WMMA GEMM,
// then softmax/top-2 -> scaled per-token coefficients C[T][64].
// ============================================================================
__global__ __launch_bounds__(256) void goat_gate_kernel(
    const float* __restrict__ x, const float* __restrict__ gW,
    const float* __restrict__ A, float* __restrict__ C)
{
  __shared__ __align__(16) unsigned char smem[40960];
  unsigned short* Xh = (unsigned short*)smem;        // 128*40
  unsigned short* Xl = Xh + 128 * LDSTR;
  unsigned short* Wh = Xl + 128 * LDSTR;             // 80*40
  unsigned short* Wl = Wh + 80 * LDSTR;
  float* GH = (float*)smem;                          // overlay: 128 x 80 staging

  const int tid  = threadIdx.x;
  const int lane = tid & 31;
  const int wv   = tid >> 5;
  const int m0   = blockIdx.x * 128;

  v8f acc[5];
  #pragma unroll
  for (int i = 0; i < 5; ++i) acc[i] = vzero8();

  for (int k0 = 0; k0 < DIN; k0 += 32) {
    __syncthreads();
    #pragma unroll
    for (int i = 0; i < 4; ++i) {
      int f = tid + 256 * i;
      int r = f >> 3, c4 = f & 7;
      float4 v = *(const float4*)(x + (size_t)(m0 + r) * DIN + k0 + c4 * 4);
      us4 h, l;
      split4(v, h, l);
      *(us4*)(Xh + r * LDSTR + c4 * 4) = h;
      *(us4*)(Xl + r * LDSTR + c4 * 4) = l;
    }
    #pragma unroll
    for (int i = 0; i < 5; ++i) {
      int f = tid + 256 * i;
      int r = f >> 4, c2 = f & 15;
      float2 v;
      if (r < 8)       v = *(const float2*)(gW + (size_t)r * DIN + k0 + c2 * 2);
      else if (r < 72) v = *(const float2*)(A + (size_t)(r - 8) * DIN + k0 + c2 * 2);
      else             v = make_float2(0.f, 0.f);
      unsigned short h0, l0, h1, l1;
      split2(v.x, h0, l0); split2(v.y, h1, l1);
      us2 h, l;
      h[0] = h0; h[1] = h1; l[0] = l0; l[1] = l1;
      *(us2*)(Wh + r * LDSTR + c2 * 2) = h;
      *(us2*)(Wl + r * LDSTR + c2 * 2) = l;
    }
    __syncthreads();

    v16bf aH = fragA(Xh, wv * 16, lane);
    v16bf aL = fragA(Xl, wv * 16, lane);
    #pragma unroll
    for (int ni = 0; ni < 5; ++ni) {
      v16bf bH = fragB(Wh, ni * 16, lane);
      v16bf bL = fragB(Wl, ni * 16, lane);
      acc[ni] = __builtin_amdgcn_wmma_f32_16x16x32_bf16(false, aH, false, bH, (short)0, acc[ni], false, false);
      acc[ni] = __builtin_amdgcn_wmma_f32_16x16x32_bf16(false, aH, false, bL, (short)0, acc[ni], false, false);
      acc[ni] = __builtin_amdgcn_wmma_f32_16x16x32_bf16(false, aL, false, bH, (short)0, acc[ni], false, false);
    }
  }

  __syncthreads();
  {
    int l2 = lane & 15, s = (lane >> 4) & 1;
    #pragma unroll
    for (int ni = 0; ni < 5; ++ni) {
      int col  = ni * 16 + l2;
      int rowb = wv * 16 + s * 8;
      #pragma unroll
      for (int v = 0; v < 8; ++v)
        GH[(rowb + v) * 80 + col] = acc[ni][v];
    }
  }
  __syncthreads();

  if (tid < 128) {
    const float* gh = GH + tid * 80;
    float p[8];
    float mx = gh[0];
    #pragma unroll
    for (int e = 1; e < 8; ++e) mx = fmaxf(mx, gh[e]);
    #pragma unroll
    for (int e = 0; e < 8; ++e) p[e] = __expf(gh[e] - mx);
    int i1 = 0;
    #pragma unroll
    for (int e = 1; e < 8; ++e) if (p[e] > p[i1]) i1 = e;
    int i2 = (i1 == 0) ? 1 : 0;
    #pragma unroll
    for (int e = 0; e < 8; ++e) if (e != i1 && p[e] > p[i2]) i2 = e;
    float inv = 1.f / (p[i1] + p[i2]);
    float w1 = p[i1] * inv, w2 = p[i2] * inv;
    float* crow = C + (size_t)(m0 + tid) * 64;
    #pragma unroll
    for (int j = 0; j < 64; ++j) {
      int e = j >> 3;
      float wgt = (e == i1) ? w1 : ((e == i2) ? w2 : 0.f);
      crow[j] = wgt * 2.0f * gh[8 + j];
    }
  }
}

// ---- shared epilogue + LoRA fp32 tail (used by both main kernels) ----------
__device__ __forceinline__ void lora_tail_and_store(
    v8f acc[2][4], const float* __restrict__ C, const float* __restrict__ B,
    float* __restrict__ out, int m0, int n0, int wm, int wn, int lane)
{
  int l2 = lane & 15, s = (lane >> 4) & 1;
  #pragma unroll
  for (int kt = 0; kt < 16; ++kt) {
    int kq = kt * 4 + s * 2;
    int e = kq >> 3, r = kq & 7;
    v2f aT[2], bT[4];
    #pragma unroll
    for (int mi = 0; mi < 2; ++mi) {
      int row = m0 + wm * 32 + mi * 16 + l2;
      aT[mi] = *(const v2f*)(C + (size_t)row * 64 + kq);
    }
    #pragma unroll
    for (int ni = 0; ni < 4; ++ni) {
      int o = n0 + wn * 64 + ni * 16 + l2;
      bT[ni] = *(const v2f*)(B + ((size_t)e * DOUT + o) * 8 + r);
    }
    #pragma unroll
    for (int mi = 0; mi < 2; ++mi)
      #pragma unroll
      for (int ni = 0; ni < 4; ++ni)
        acc[mi][ni] = __builtin_amdgcn_wmma_f32_16x16x4_f32(false, aT[mi], false, bT[ni], (short)0, acc[mi][ni], false, false);
  }
  #pragma unroll
  for (int mi = 0; mi < 2; ++mi)
    #pragma unroll
    for (int ni = 0; ni < 4; ++ni) {
      int rowb = m0 + wm * 32 + mi * 16 + s * 8;
      int col  = n0 + wn * 64 + ni * 16 + l2;
      #pragma unroll
      for (int v = 0; v < 8; ++v)
        out[(size_t)(rowb + v) * DOUT + col] = acc[mi][ni][v];
    }
}

// ============================================================================
// Kernel 2a (preferred): bf16x3 GEMM streaming pre-split planes via
// global_load_async_to_lds_b128 with a 3-stage software pipeline.
// Dynamic LDS: 3 stages x 40960 B = 122880 B.
// ============================================================================
__global__ __launch_bounds__(256) void goat_main_pre(
    const unsigned short* __restrict__ Xhp, const unsigned short* __restrict__ Xlp,
    const unsigned short* __restrict__ Whp, const unsigned short* __restrict__ Wlp,
    const float* __restrict__ C, const float* __restrict__ B,
    float* __restrict__ out)
{
  extern __shared__ unsigned char DYN[];   // 3 * TILE_B

  const int tid  = threadIdx.x;
  const int lane = tid & 31;
  const int wv   = tid >> 5;
  const int wm   = wv & 3;
  const int wn   = wv >> 2;
  const int m0   = blockIdx.y * 128;
  const int n0   = blockIdx.x * 128;

  // Per-lane constants: wave wv fills tile rows [wv*16, wv*16+16) of each
  // plane; 2 issues/plane, 8 rows/issue, 4 lanes per 64B row (16B each).
  const unsigned ldsBase = (unsigned)(uintptr_t)(void*)DYN;  // low 32 = LDS off
  unsigned voff[2], lrow[2];
  #pragma unroll
  for (int i = 0; i < 2; ++i) {
    int rloc = wv * 16 + i * 8 + (lane >> 2);
    int colb = (lane & 3) * 16;
    voff[i] = (unsigned)rloc * (DIN * 2) + colb;
    lrow[i] = ldsBase + (unsigned)rloc * (LDSTR * 2) + colb;
  }
  const unsigned char* sXh = (const unsigned char*)Xhp + (size_t)m0 * DIN * 2;
  const unsigned char* sXl = (const unsigned char*)Xlp + (size_t)m0 * DIN * 2;
  const unsigned char* sWh = (const unsigned char*)Whp + (size_t)n0 * DIN * 2;
  const unsigned char* sWl = (const unsigned char*)Wlp + (size_t)n0 * DIN * 2;

  auto issue_tile = [&](int kk, int b) {
    size_t kb = (size_t)kk * 2;                  // 64 B per K-step
    unsigned bo = (unsigned)b * (unsigned)TILE_B;
    #pragma unroll
    for (int i = 0; i < 2; ++i) {
      async_ld_b128(lrow[i] + bo,                          voff[i], sXh + kb);
      async_ld_b128(lrow[i] + bo + 1 * PLANE_USH * 2,      voff[i], sXl + kb);
      async_ld_b128(lrow[i] + bo + 2 * PLANE_USH * 2,      voff[i], sWh + kb);
      async_ld_b128(lrow[i] + bo + 3 * PLANE_USH * 2,      voff[i], sWl + kb);
    }
  };

  v8f acc[2][4];
  #pragma unroll
  for (int mi = 0; mi < 2; ++mi)
    #pragma unroll
    for (int ni = 0; ni < 4; ++ni) acc[mi][ni] = vzero8();

  // Prologue: stage tiles 0 and 1.
  issue_tile(0, 0);
  issue_tile(32, 1);

  int b = 0;
  for (int k0 = 0; k0 < DIN; k0 += 32) {
    __syncthreads();   // everyone done reading the buffer tile k+2 will use
    int rem = DIN - k0;
    if (rem > 64) {                 // tile k+2 exists
      int b2 = b + 2; if (b2 >= 3) b2 -= 3;
      issue_tile(k0 + 64, b2);
      wait_async16();               // <=16 left => tile k's 8 copies done
    } else if (rem > 32) {
      wait_async8();                // only tile k+1 still in flight
    } else {
      wait_async0();
    }
    __syncthreads();                // tile k visible to all waves

    const unsigned short* Xh = (const unsigned short*)(DYN + b * TILE_B);
    const unsigned short* Xl = Xh + PLANE_USH;
    const unsigned short* Wh = Xh + 2 * PLANE_USH;
    const unsigned short* Wl = Xh + 3 * PLANE_USH;

    v16bf aH[2], aL[2], bH[4], bL[4];
    #pragma unroll
    for (int mi = 0; mi < 2; ++mi) {
      aH[mi] = fragA(Xh, wm * 32 + mi * 16, lane);
      aL[mi] = fragA(Xl, wm * 32 + mi * 16, lane);
    }
    #pragma unroll
    for (int ni = 0; ni < 4; ++ni) {
      bH[ni] = fragB(Wh, wn * 64 + ni * 16, lane);
      bL[ni] = fragB(Wl, wn * 64 + ni * 16, lane);
    }
    #pragma unroll
    for (int mi = 0; mi < 2; ++mi)
      #pragma unroll
      for (int ni = 0; ni < 4; ++ni) {
        acc[mi][ni] = __builtin_amdgcn_wmma_f32_16x16x32_bf16(false, aH[mi], false, bH[ni], (short)0, acc[mi][ni], false, false);
        acc[mi][ni] = __builtin_amdgcn_wmma_f32_16x16x32_bf16(false, aH[mi], false, bL[ni], (short)0, acc[mi][ni], false, false);
        acc[mi][ni] = __builtin_amdgcn_wmma_f32_16x16x32_bf16(false, aL[mi], false, bH[ni], (short)0, acc[mi][ni], false, false);
      }

    if (++b >= 3) b = 0;
  }

  lora_tail_and_store(acc, C, B, out, m0, n0, wm, wn, lane);
}

// ============================================================================
// Kernel 2b (fallback, small ws): fused-conversion variant.
// ============================================================================
__global__ __launch_bounds__(256) void goat_main_fused(
    const float* __restrict__ x, const float* __restrict__ Wb,
    const float* __restrict__ C, const float* __restrict__ B,
    float* __restrict__ out)
{
  __shared__ __align__(16) unsigned short Xh[PLANE_USH];
  __shared__ __align__(16) unsigned short Xl[PLANE_USH];
  __shared__ __align__(16) unsigned short Wh[PLANE_USH];
  __shared__ __align__(16) unsigned short Wl[PLANE_USH];

  const int tid  = threadIdx.x;
  const int lane = tid & 31;
  const int wv   = tid >> 5;
  const int wm   = wv & 3;
  const int wn   = wv >> 2;
  const int m0   = blockIdx.y * 128;
  const int n0   = blockIdx.x * 128;

  v8f acc[2][4];
  #pragma unroll
  for (int mi = 0; mi < 2; ++mi)
    #pragma unroll
    for (int ni = 0; ni < 4; ++ni) acc[mi][ni] = vzero8();

  for (int k0 = 0; k0 < DIN; k0 += 32) {
    __syncthreads();
    #pragma unroll
    for (int i = 0; i < 4; ++i) {
      int f = tid + 256 * i;
      int r = f >> 3, c4 = f & 7;
      float4 vx = *(const float4*)(x  + (size_t)(m0 + r) * DIN + k0 + c4 * 4);
      float4 vw = *(const float4*)(Wb + (size_t)(n0 + r) * DIN + k0 + c4 * 4);
      us4 h, l;
      split4(vx, h, l);
      *(us4*)(Xh + r * LDSTR + c4 * 4) = h;
      *(us4*)(Xl + r * LDSTR + c4 * 4) = l;
      split4(vw, h, l);
      *(us4*)(Wh + r * LDSTR + c4 * 4) = h;
      *(us4*)(Wl + r * LDSTR + c4 * 4) = l;
    }
    __syncthreads();

    v16bf aH[2], aL[2], bH[4], bL[4];
    #pragma unroll
    for (int mi = 0; mi < 2; ++mi) {
      aH[mi] = fragA(Xh, wm * 32 + mi * 16, lane);
      aL[mi] = fragA(Xl, wm * 32 + mi * 16, lane);
    }
    #pragma unroll
    for (int ni = 0; ni < 4; ++ni) {
      bH[ni] = fragB(Wh, wn * 64 + ni * 16, lane);
      bL[ni] = fragB(Wl, wn * 64 + ni * 16, lane);
    }
    #pragma unroll
    for (int mi = 0; mi < 2; ++mi)
      #pragma unroll
      for (int ni = 0; ni < 4; ++ni) {
        acc[mi][ni] = __builtin_amdgcn_wmma_f32_16x16x32_bf16(false, aH[mi], false, bH[ni], (short)0, acc[mi][ni], false, false);
        acc[mi][ni] = __builtin_amdgcn_wmma_f32_16x16x32_bf16(false, aH[mi], false, bL[ni], (short)0, acc[mi][ni], false, false);
        acc[mi][ni] = __builtin_amdgcn_wmma_f32_16x16x32_bf16(false, aL[mi], false, bH[ni], (short)0, acc[mi][ni], false, false);
      }
  }

  lora_tail_and_store(acc, C, B, out, m0, n0, wm, wn, lane);
}

extern "C" void kernel_launch(void* const* d_in, const int* in_sizes, int n_in,
                              void* d_out, int out_size, void* d_ws, size_t ws_size,
                              hipStream_t stream) {
  const float* x  = (const float*)d_in[0];   // [T, 4096]
  const float* Wb = (const float*)d_in[1];   // [4096, 4096]
  const float* gW = (const float*)d_in[2];   // [8, 4096]
  const float* A  = (const float*)d_in[3];   // [8, 8, 4096] == [64, 4096]
  const float* B  = (const float*)d_in[4];   // [8, 4096, 8]
  float* out = (float*)d_out;
  const int T = in_sizes[0] / DIN;           // 8192

  unsigned char* ws = (unsigned char*)d_ws;
  float* C = (float*)ws;                     // [T, 64]
  size_t cBytes = (size_t)T * 64 * sizeof(float);
  size_t xPlane = (size_t)T * DIN * 2;       // bf16 plane bytes
  size_t wPlane = (size_t)DOUT * DIN * 2;
  size_t need   = cBytes + 2 * xPlane + 2 * wPlane;

  goat_gate_kernel<<<dim3(T / 128), dim3(256), 0, stream>>>(x, gW, A, C);

  if (ws_size >= need) {
    unsigned short* Xhp = (unsigned short*)(ws + cBytes);
    unsigned short* Xlp = (unsigned short*)(ws + cBytes + xPlane);
    unsigned short* Whp = (unsigned short*)(ws + cBytes + 2 * xPlane);
    unsigned short* Wlp = (unsigned short*)(ws + cBytes + 2 * xPlane + wPlane);
    int xN4 = (T * DIN) / 4, wN4 = (DOUT * DIN) / 4;
    split_planes_kernel<<<dim3((xN4 + 255) / 256), dim3(256), 0, stream>>>(x,  Xhp, Xlp, xN4);
    split_planes_kernel<<<dim3((wN4 + 255) / 256), dim3(256), 0, stream>>>(Wb, Whp, Wlp, wN4);
    goat_main_pre<<<dim3(DOUT / 128, T / 128), dim3(256), 3 * TILE_B, stream>>>(Xhp, Xlp, Whp, Wlp, C, B, out);
  } else {
    goat_main_fused<<<dim3(DOUT / 128, T / 128), dim3(256), 0, stream>>>(x, Wb, C, B, out);
  }
  (void)n_in; (void)out_size;
}
